// AdditiveAttention_3633542333093
// MI455X (gfx1250) — matrix operand order repaired and checked by reading, not compiled, not run on hardware
//
#include <hip/hip_runtime.h>
#include <hip/hip_bf16.h>

#define NB 16      // batch
#define NS 8192    // sequence
#define NH 256     // hidden

typedef __attribute__((ext_vector_type(16))) _Float16 v16h;
typedef __attribute__((ext_vector_type(8)))  float    v8f;
typedef __attribute__((ext_vector_type(4)))  float    v4f;

#if __has_builtin(__builtin_amdgcn_tanhf)
#define TANHF(x) __builtin_amdgcn_tanhf(x)
#else
#define TANHF(x) tanhf(x)
#endif

// ---------------------------------------------------------------------------
// K0: pack Wk = W_attn[:, H:] (row-major [o][h]) into per-lane WMMA B-fragment
// order (f16). Layout: [ks(8)][t(16)][lane(32)][e(16)] halves, where for
// lane l: N = t*16 + (l&15), half-group sub = l>>4, and K = ks*32 + sub*8 + e
// for e < 8, K = ks*32 + 16 + sub*8 + (e-8) for e >= 8.
// Each K-step is then a contiguous 16KB block -> linear async copy to LDS.
// ---------------------------------------------------------------------------
__global__ __launch_bounds__(256) void pack_wk_kernel(
    const float* __restrict__ W, _Float16* __restrict__ wkp) {
  int idx  = blockIdx.x * 256 + threadIdx.x;   // 0..65535
  int e    = idx & 15;
  int lane = (idx >> 4) & 31;
  int t    = (idx >> 9) & 15;
  int ks   = idx >> 13;
  int o    = t * 16 + (lane & 15);
  int sub  = lane >> 4;
  int k    = ks * 32 + (sub << 3) + (e & 7) + ((e >> 3) << 4);
  wkp[idx] = (_Float16)W[o * (2 * NH) + NH + k];
}

// ---------------------------------------------------------------------------
// K1: qb[b][o] = query[b]·Wq[o] + b_attn[o]
// ---------------------------------------------------------------------------
__global__ __launch_bounds__(256) void qproj_kernel(
    const float* __restrict__ query, const float* __restrict__ W,
    const float* __restrict__ b_attn, float* __restrict__ qb) {
  int b = blockIdx.x;
  int o = threadIdx.x;
  const float* q = query + b * NH;
  const float* w = W + o * (2 * NH);
  float acc = b_attn[o];
  #pragma unroll 8
  for (int h = 0; h < NH; ++h) acc += q[h] * w[h];
  qb[b * NH + o] = acc;
}

// ---------------------------------------------------------------------------
// K2: scores[b][s] = sum_o tanh( (K Wk^T)[s,o] + qb[b][o] ) * v[o]
// One wave per 16-row tile, 8 waves/WG.
//  - Full A-tile (K=256) preloaded fp32 -> f16 up front (HBM stream starts
//    immediately, never blocked by barriers).
//  - B fragments double-buffer staged into LDS via
//    global_load_async_to_lds_b128 with one full iteration of lead time.
//  - Inner loop: register double-buffered ds_load_b128 + 16 v_wmma per step.
//  - Fused tanh + dot(v) epilogue with half-wave shuffle reduction.
// ---------------------------------------------------------------------------
__global__ __launch_bounds__(256) void scores_kernel(
    const float*    __restrict__ key,    // [NB][NS][NH] fp32
    const _Float16* __restrict__ wkp,    // packed B fragments [8][16KB]
    const float*    __restrict__ qb,     // [NB][NH]
    const float*    __restrict__ vvec,   // [NH]
    float*          __restrict__ scores) // [NB][NS]
{
  __shared__ _Float16 bstage[2][8192];   // 2 x 16KB double buffer

  const int tid  = threadIdx.x;
  const int wave = tid >> 5;               // 0..7
  const int lane = tid & 31;
  const int b    = blockIdx.x >> 6;        // NS/128 = 64 tiles per batch
  const int tile = blockIdx.x & 63;
  const int s0   = tile * 128 + wave * 16; // 16 rows for this wave
  const int m    = lane & 15;              // row within tile
  const int sub  = lane >> 4;

  // A-fragment source: row s0+m, two 8-float runs per K-step (ISA A layout)
  const float* arow = key + ((size_t)b * NS + s0 + m) * NH + sub * 8;

  // generic->LDS: low 32 bits of a shared-memory generic address are the
  // LDS byte offset (aperture truncation rule)
  const unsigned lds0 = (unsigned)(uintptr_t)(&bstage[0][0]) + tid * 16u;
  const char*    gsrc = (const char*)wkp + tid * 16;

  // issue async stage for K-step ks into buffer buf: 4 x b128 per thread
  #define ISSUE_STAGE(ks, buf)                                               \
    {                                                                        \
      const char* g = gsrc + (ks) * 16384;                                   \
      unsigned    l = lds0 + (buf) * 16384u;                                 \
      _Pragma("unroll")                                                      \
      for (int r = 0; r < 4; ++r) {                                          \
        asm volatile("global_load_async_to_lds_b128 %0, %1, off"             \
                     :: "v"(l + r * 4096u), "v"(g + r * 4096) : "memory");   \
      }                                                                      \
    }

  // Kick off the first two LDS stages before anything else.
  ISSUE_STAGE(0, 0);
  ISSUE_STAGE(1, 1);

  // Preload the whole A tile (K=256) and convert fp32 -> f16: 8 x v16h.
  v16h afrag[8];
  #pragma unroll
  for (int ks = 0; ks < 8; ++ks) {
    const v4f a0 = *(const v4f*)(arow + ks * 32 + 0);
    const v4f a1 = *(const v4f*)(arow + ks * 32 + 4);
    const v4f a2 = *(const v4f*)(arow + ks * 32 + 16);
    const v4f a3 = *(const v4f*)(arow + ks * 32 + 20);
    #pragma unroll
    for (int i = 0; i < 4; ++i) {
      afrag[ks][i]      = (_Float16)a0[i];
      afrag[ks][4 + i]  = (_Float16)a1[i];
      afrag[ks][8 + i]  = (_Float16)a2[i];
      afrag[ks][12 + i] = (_Float16)a3[i];
    }
  }

  v8f acc[16] = {};

  for (int ks = 0; ks < 8; ++ks) {
    if (ks < 7) asm volatile("s_wait_asynccnt 4" ::: "memory"); // stage ks landed
    else        asm volatile("s_wait_asynccnt 0" ::: "memory");
    __syncthreads();                                   // all waves' stage visible

    const v16h a = afrag[ks];
    const _Float16* bbuf = &bstage[ks & 1][lane * 16];

    // register double-buffered B fragments: load t+1 before WMMA t
    v16h bf = *(const v16h*)(bbuf);
    #pragma unroll
    for (int t = 0; t < 16; ++t) {
      v16h bfn = bf;
      if (t < 15) bfn = *(const v16h*)(bbuf + (t + 1) * 512);
      acc[t] = __builtin_amdgcn_wmma_f32_16x16x32_f16(
          false, a, false, bf, (short)0, acc[t], false, false);
      bf = bfn;
    }
    __syncthreads();                                   // buffer reuse fence
    if (ks < 6) ISSUE_STAGE(ks + 2, ks & 1);           // refill freed buffer
  }

  // Epilogue: lane holds C elements (M = r + 8*sub, N = t*16 + (lane&15)).
  float p[8];
  #pragma unroll
  for (int r = 0; r < 8; ++r) p[r] = 0.f;
  const int nb = lane & 15;
  #pragma unroll
  for (int t = 0; t < 16; ++t) {
    const int n = t * 16 + nb;
    const float qv = qb[b * NH + n];
    const float vv = vvec[n];
    #pragma unroll
    for (int r = 0; r < 8; ++r) p[r] += TANHF(acc[t][r] + qv) * vv;
  }
  // reduce over the 16 lanes of each half-wave (xor 1,2,4,8 stays in-half)
  #pragma unroll
  for (int r = 0; r < 8; ++r) {
    #pragma unroll
    for (int off = 1; off < 16; off <<= 1)
      p[r] += __shfl_xor(p[r], off, 32);
  }
  if (nb == 0) {
    #pragma unroll
    for (int r = 0; r < 8; ++r)
      scores[(size_t)b * NS + s0 + r + sub * 8] = p[r];
  }
}

// ---------------------------------------------------------------------------
// K3: softmax over S per batch; writes attn weights directly to d_out.
// ---------------------------------------------------------------------------
__global__ __launch_bounds__(256) void softmax_kernel(
    const float* __restrict__ scores, float* __restrict__ attn) {
  const int b = blockIdx.x;
  const int tid = threadIdx.x;
  __shared__ float red[8];
  const float* srow = scores + (size_t)b * NS;

  float mx = -3.402823466e38f;
  for (int i = tid; i < NS; i += 256) mx = fmaxf(mx, srow[i]);
  #pragma unroll
  for (int off = 16; off > 0; off >>= 1) mx = fmaxf(mx, __shfl_xor(mx, off, 32));
  if ((tid & 31) == 0) red[tid >> 5] = mx;
  __syncthreads();
  float gmax = red[0];
  #pragma unroll
  for (int i = 1; i < 8; ++i) gmax = fmaxf(gmax, red[i]);
  __syncthreads();

  float sum = 0.f;
  for (int i = tid; i < NS; i += 256) sum += __expf(srow[i] - gmax);
  #pragma unroll
  for (int off = 16; off > 0; off >>= 1) sum += __shfl_xor(sum, off, 32);
  if ((tid & 31) == 0) red[tid >> 5] = sum;
  __syncthreads();
  float tot = 0.f;
  #pragma unroll
  for (int i = 0; i < 8; ++i) tot += red[i];
  const float inv = 1.0f / tot;

  for (int i = tid; i < NS; i += 256)
    attn[(size_t)b * NS + i] = __expf(srow[i] - gmax) * inv;
}

// ---------------------------------------------------------------------------
// K4: partial context: chunk of 256 s-values per WG, coalesced over h.
// ---------------------------------------------------------------------------
__global__ __launch_bounds__(256) void ctx_partial_kernel(
    const float* __restrict__ attn, const float* __restrict__ value,
    float* __restrict__ partials) {
  const int b = blockIdx.x >> 5;   // 32 chunks per batch
  const int c = blockIdx.x & 31;
  const int h = threadIdx.x;
  const int sbase = c * (NS / 32); // 256 s per chunk
  const float* val = value + ((size_t)b * NS + sbase) * NH + h;
  const float* w = attn + (size_t)b * NS + sbase;
  float acc = 0.f;
  #pragma unroll 4
  for (int s = 0; s < NS / 32; ++s) acc += w[s] * val[(size_t)s * NH];
  partials[((size_t)b * 32 + c) * NH + h] = acc;
}

// ---------------------------------------------------------------------------
// K5: deterministic reduction of 32 partials -> context in d_out.
// ---------------------------------------------------------------------------
__global__ __launch_bounds__(256) void ctx_reduce_kernel(
    const float* __restrict__ partials, float* __restrict__ ctx) {
  const int b = blockIdx.x;
  const int h = threadIdx.x;
  float acc = 0.f;
  #pragma unroll
  for (int c = 0; c < 32; ++c) acc += partials[((size_t)b * 32 + c) * NH + h];
  ctx[b * NH + h] = acc;
}

// ---------------------------------------------------------------------------
extern "C" void kernel_launch(void* const* d_in, const int* in_sizes, int n_in,
                              void* d_out, int out_size, void* d_ws, size_t ws_size,
                              hipStream_t stream) {
  const float* query  = (const float*)d_in[0]; // [16][1][256]
  const float* key    = (const float*)d_in[1]; // [16][8192][256]
  const float* value  = (const float*)d_in[2]; // [16][8192][256]
  const float* W_attn = (const float*)d_in[3]; // [256][512]
  const float* b_attn = (const float*)d_in[4]; // [256]
  const float* vvec   = (const float*)d_in[5]; // [256]

  float* out  = (float*)d_out;
  float* ctx  = out;               // B*H = 4096 floats (context)
  float* attn = out + NB * NH;     // B*S = 131072 floats (attn weights)

  float* ws       = (float*)d_ws;
  float* scores   = ws;                                  // 131072 floats
  float* qb       = ws + (size_t)NB * NS;                // 4096 floats
  _Float16* wkp   = (_Float16*)(ws + (size_t)NB * NS + NB * NH); // 65536 halves
  float* partials = ws + (size_t)NB * NS + NB * NH + 32768;      // 131072 floats

  pack_wk_kernel    <<<256, 256, 0, stream>>>(W_attn, wkp);
  qproj_kernel      <<<NB, NH, 0, stream>>>(query, W_attn, b_attn, qb);
  scores_kernel     <<<NB * 64, 256, 0, stream>>>(key, wkp, qb, vvec, scores);
  softmax_kernel    <<<NB, 256, 0, stream>>>(scores, attn);
  ctx_partial_kernel<<<NB * 32, NH, 0, stream>>>(attn, value, partials);
  ctx_reduce_kernel <<<NB, NH, 0, stream>>>(partials, ctx);
}
